// ImprovedConvKNRM_41686952575648
// MI455X (gfx1250) — compile-verified, matrix operand order checked by reading, not run on hardware
//
#include <hip/hip_runtime.h>
#include <hip/hip_bf16.h>
#include <math.h>

typedef __attribute__((ext_vector_type(16))) _Float16 v16h;
typedef __attribute__((ext_vector_type(8)))  _Float16 v8h;
typedef __attribute__((ext_vector_type(8)))  float    v8f;

union H16 { v16h v; v8h h[2]; };

// ---------------------------------------------------------------------------
// Weight repack: (C=128, E=128, 3) f32  ->  Wt[c][kk*128 + e] f16 (128 x 384)
// Makes the WMMA B-fragment a plain row-major 16x32 sub-tile of Wt.
// ---------------------------------------------------------------------------
__global__ void cvt_weights_kernel(const float* __restrict__ W,
                                   _Float16* __restrict__ Wt) {
  int i = blockIdx.x * blockDim.x + threadIdx.x;
  if (i >= 128 * 384) return;
  int c = i / 384, j = i % 384;
  int kk = j / 128, e = j % 128;
  Wt[i] = (_Float16)W[(size_t)(c * 128 + e) * 3 + kk];
}

// ---------------------------------------------------------------------------
// Fused: embedding gather (f32->f16) + Conv1d(k=3,SAME) + bias + ReLU +
// L2-normalize over C, output f16 row-major (B*L, 128).
// One block (8 waves) = one batch x 128 positions; wave w owns 16 positions.
// GEMM: (16 pos) x (K=384 window feats) x (128 channels) via 96 WMMAs/wave.
// ---------------------------------------------------------------------------
__global__ __launch_bounds__(256) void conv_norm_kernel(
    const int* __restrict__ tok, const float* __restrict__ emb,
    const _Float16* __restrict__ Wt, const float* __restrict__ bias,
    _Float16* __restrict__ outp, int L, int tpb) {
  __shared__ __align__(16) unsigned int xw32[130 * 64];  // 130 rows x 128 f16

  const int b    = blockIdx.x / tpb;
  const int base = (blockIdx.x % tpb) * 128;
  const int tid  = threadIdx.x;
  const int lane = tid & 31, wave = tid >> 5;
  const int lm   = lane & 15;           // A/B fragment row within tile
  const int hi8  = (lane >> 4) * 8;     // K sub-block select per ISA layout
  const int* trow = tok + (size_t)b * L;

  // Stage token-window embeddings into LDS as f16 (zero-pad the SAME halo).
  for (int i = tid; i < 130 * 64; i += 256) {
    int row = i >> 6, col = i & 63;
    int p = base - 1 + row;
    unsigned int v = 0u;
    if (p >= 0 && p < L) {
      const float* ep = emb + (size_t)trow[p] * 128 + col * 2;
      union { _Float16 h[2]; unsigned int u; } cv;
      cv.h[0] = (_Float16)ep[0];
      cv.h[1] = (_Float16)ep[1];
      v = cv.u;
    }
    xw32[i] = v;
  }
  __syncthreads();

  const _Float16* xw = (const _Float16*)xw32;

  v8f acc[8];
  v8f z = {};
#pragma unroll
  for (int nt = 0; nt < 8; ++nt) acc[nt] = z;

  // K = 384 = 3 taps x 128 feats, 12 chunks of 32.
#pragma unroll
  for (int cc = 0; cc < 12; ++cc) {
    const int kk = cc >> 2, e0 = (cc & 3) * 32;
    H16 a;
    const _Float16* ap = xw + (size_t)(wave * 16 + lm + kk) * 128 + e0 + hi8;
    a.h[0] = *(const v8h*)ap;
    a.h[1] = *(const v8h*)(ap + 16);
#pragma unroll
    for (int nt = 0; nt < 8; ++nt) {  // 8 tiles of 16 output channels
      const _Float16* bp =
          Wt + (size_t)(nt * 16 + lm) * 384 + kk * 128 + e0 + hi8;
      H16 bf;
      bf.h[0] = *(const v8h*)bp;
      bf.h[1] = *(const v8h*)(bp + 16);
      acc[nt] = __builtin_amdgcn_wmma_f32_16x16x32_f16(
          false, a.v, false, bf.v, (short)0, acc[nt], false, false);
    }
  }

  // Epilogue: bias + ReLU, L2 norm over 128 channels (cross-lane over the 16
  // lanes of each half-wave that hold a row's columns), scale, store f16.
  float bs[8];
#pragma unroll
  for (int nt = 0; nt < 8; ++nt) bs[nt] = bias[nt * 16 + lm];

  float ss[8];
#pragma unroll
  for (int r = 0; r < 8; ++r) {
    float s = 0.f;
#pragma unroll
    for (int nt = 0; nt < 8; ++nt) {
      float y = fmaxf(acc[nt][r] + bs[nt], 0.f);
      acc[nt][r] = y;
      s += y * y;
    }
    ss[r] = s;
  }
#pragma unroll
  for (int m = 1; m < 16; m <<= 1) {
#pragma unroll
    for (int r = 0; r < 8; ++r) ss[r] += __shfl_xor(ss[r], m);
  }
#pragma unroll
  for (int r = 0; r < 8; ++r) ss[r] = 1.0f / fmaxf(sqrtf(ss[r]), 1e-12f);

  const size_t rowbase = (size_t)b * L + base + wave * 16 + (lane >> 4) * 8;
#pragma unroll
  for (int r = 0; r < 8; ++r) {
    _Float16* op = outp + (rowbase + r) * 128 + lm;
#pragma unroll
    for (int nt = 0; nt < 8; ++nt) op[nt * 16] = (_Float16)(acc[nt][r] * ss[r]);
  }
}

// ---------------------------------------------------------------------------
// Fused cosine-sim GEMM + Gaussian RBF kernel pooling.
// Block = one batch (8 waves); wave w owns query tile w (16 rows).
// Loop over 128 doc tiles: 4 WMMAs (K=128) -> 16x16 sim tile in regs ->
// accumulate 11 exp kernels per element (sum over doc). The Gaussian exponent
// is expanded to a quadratic a*s^2 + b*s + c so the hot loop is 2 FMAs + one
// trans-unit exp per (row, kernel). Cross-lane reduce, then max over the 128
// query rows -> kf[b][11].
// ---------------------------------------------------------------------------
__global__ __launch_bounds__(256) void sim_pool_kernel(
    const _Float16* __restrict__ qn, const _Float16* __restrict__ dn,
    const float* __restrict__ mu, const float* __restrict__ sigma,
    float* __restrict__ kf) {
  __shared__ float kq[128 * 11];

  const int b    = blockIdx.x;
  const int tid  = threadIdx.x;
  const int lane = tid & 31, wave = tid >> 5;
  const int lm   = lane & 15;
  const int hi8  = (lane >> 4) * 8;

  // w*(s-mu)^2 == ak*s^2 + bk*s + ck  with  w = -1/(2*sigma^2)
  float ak[11], bk[11], ck[11];
#pragma unroll
  for (int k = 0; k < 11; ++k) {
    float m = mu[k];
    float s = sigma[k];
    float w = -0.5f / (s * s);
    ak[k] = w;
    bk[k] = -2.0f * w * m;
    ck[k] = w * m * m;
  }

  // A fragments (query tile), kept live across the whole doc loop.
  H16 aq[4];
  const _Float16* qrow = qn + ((size_t)b * 128 + wave * 16 + lm) * 128;
#pragma unroll
  for (int c = 0; c < 4; ++c) {
    const _Float16* ap = qrow + c * 32 + hi8;
    aq[c].h[0] = *(const v8h*)ap;
    aq[c].h[1] = *(const v8h*)(ap + 16);
  }

  float pk[8][11];
#pragma unroll
  for (int r = 0; r < 8; ++r)
#pragma unroll
    for (int k = 0; k < 11; ++k) pk[r][k] = 0.f;

  const _Float16* dbase = dn + ((size_t)b * 2048 + lm) * 128;

  for (int dt = 0; dt < 128; ++dt) {
    const _Float16* drow = dbase + (size_t)dt * 16 * 128;

    // Prefetch the doc tile two iterations ahead (global_prefetch_b8).
    if (dt + 2 < 128)
      __builtin_prefetch(drow + (size_t)2 * 16 * 128, 0, 1);

    v8f acc = {};
#pragma unroll
    for (int c = 0; c < 4; ++c) {
      const _Float16* bp = drow + c * 32 + hi8;
      H16 bf;
      bf.h[0] = *(const v8h*)bp;
      bf.h[1] = *(const v8h*)(bp + 16);
      acc = __builtin_amdgcn_wmma_f32_16x16x32_f16(
          false, aq[c].v, false, bf.v, (short)0, acc, false, false);
    }
#pragma unroll
    for (int r = 0; r < 8; ++r) {
      float s  = acc[r];
      float s2 = s * s;
#pragma unroll
      for (int k = 0; k < 11; ++k) {
        pk[r][k] += __expf(fmaf(ak[k], s2, fmaf(bk[k], s, ck[k])));
      }
    }
  }

  // Sum over the 16 doc-columns held by each half-wave.
#pragma unroll
  for (int m = 1; m < 16; m <<= 1) {
#pragma unroll
    for (int r = 0; r < 8; ++r)
#pragma unroll
      for (int k = 0; k < 11; ++k) pk[r][k] += __shfl_xor(pk[r][k], m);
  }

  if (lm == 0) {
    int q0 = wave * 16 + (lane >> 4) * 8;
#pragma unroll
    for (int r = 0; r < 8; ++r)
#pragma unroll
      for (int k = 0; k < 11; ++k) kq[(q0 + r) * 11 + k] = pk[r][k];
  }
  __syncthreads();

  if (tid < 11) {
    float m = -1e30f;
    for (int q = 0; q < 128; ++q) m = fmaxf(m, kq[q * 11 + tid]);
    kf[b * 11 + tid] = m;
  }
}

// ---------------------------------------------------------------------------
// Tiny MLP head: h = tanh(kf @ Wd^T + bd); out = 2*sigmoid(h @ Wo^T + bo)
// ---------------------------------------------------------------------------
__global__ void final_mlp_kernel(const float* __restrict__ kf,
                                 const float* __restrict__ dw,
                                 const float* __restrict__ db,
                                 const float* __restrict__ ow,
                                 const float* __restrict__ ob,
                                 float* __restrict__ out) {
  int b = threadIdx.x;
  if (b >= 64) return;
  float kl[11];
#pragma unroll
  for (int k = 0; k < 11; ++k) kl[k] = kf[b * 11 + k];
  float o = ob[0];
  for (int j = 0; j < 64; ++j) {
    float s = db[j];
#pragma unroll
    for (int k = 0; k < 11; ++k) s += kl[k] * dw[j * 11 + k];
    o += tanhf(s) * ow[j];
  }
  out[b] = 2.0f * (1.0f / (1.0f + __expf(-o)));
}

// ---------------------------------------------------------------------------
extern "C" void kernel_launch(void* const* d_in, const int* in_sizes, int n_in,
                              void* d_out, int out_size, void* d_ws,
                              size_t ws_size, hipStream_t stream) {
  const int*   query  = (const int*)d_in[0];    // (64, 128)
  const int*   doc    = (const int*)d_in[1];    // (64, 2048)
  const float* emb    = (const float*)d_in[2];  // (50000, 128)
  const float* qW     = (const float*)d_in[3];  // (128, 128, 3)
  const float* qb     = (const float*)d_in[4];  // (128,)
  const float* dW     = (const float*)d_in[5];  // (128, 128, 3)
  const float* db     = (const float*)d_in[6];  // (128,)
  const float* mu     = (const float*)d_in[7];  // (11,)
  const float* sigma  = (const float*)d_in[8];  // (11,)
  const float* denseW = (const float*)d_in[9];  // (64, 11)
  const float* denseb = (const float*)d_in[10]; // (64,)
  const float* outW   = (const float*)d_in[11]; // (1, 64)
  const float* outb   = (const float*)d_in[12]; // (1,)

  char* ws = (char*)d_ws;
  size_t off = 0;
  auto carve = [&](size_t bytes) {
    char* p = ws + off;
    off += (bytes + 255) & ~(size_t)255;
    return p;
  };
  _Float16* Wq  = (_Float16*)carve(128 * 384 * sizeof(_Float16));
  _Float16* Wd  = (_Float16*)carve(128 * 384 * sizeof(_Float16));
  _Float16* qnb = (_Float16*)carve((size_t)64 * 128 * 128 * sizeof(_Float16));
  _Float16* dnb = (_Float16*)carve((size_t)64 * 2048 * 128 * sizeof(_Float16));
  float*    kfb = (float*)carve(64 * 11 * sizeof(float));

  cvt_weights_kernel<<<(128 * 384 + 255) / 256, 256, 0, stream>>>(qW, Wq);
  cvt_weights_kernel<<<(128 * 384 + 255) / 256, 256, 0, stream>>>(dW, Wd);

  // Query: 64 blocks (1 tile of 128 positions each). Doc: 64*16 blocks.
  conv_norm_kernel<<<64 * 1, 256, 0, stream>>>(query, emb, Wq, qb, qnb, 128, 1);
  conv_norm_kernel<<<64 * 16, 256, 0, stream>>>(doc, emb, Wd, db, dnb, 2048, 16);

  sim_pool_kernel<<<64, 256, 0, stream>>>(qnb, dnb, mu, sigma, kfb);

  final_mlp_kernel<<<1, 64, 0, stream>>>(kfb, denseW, denseb, outW, outb,
                                         (float*)d_out);
}
